// SchNet_19018115186811
// MI455X (gfx1250) — matrix operand order repaired and checked by reading, not compiled, not run on hardware
//
#include <hip/hip_runtime.h>
#include <hip/hip_bf16.h>
#include <math.h>

typedef _Float16 f16;
typedef _Float16 v16h __attribute__((ext_vector_type(16)));
typedef float    v8f  __attribute__((ext_vector_type(8)));

#define LDA 34    // A-tile lds stride (32-k chunk + pad) -> 68B lane stride, conflict-free
#define LDB 40    // B-tile lds stride -> 16B aligned frags, conflict-free
#define LDH 132   // full-K (128) A-tile stride + pad

__device__ __forceinline__ float sspf(float v) {
  float sp = (v > 20.f) ? v : log1pf(__expf(v));
  return sp - 0.69314718056f;
}
__device__ __forceinline__ float siluf(float v) {
  return v / (1.f + __expf(-v));
}
__device__ __forceinline__ v8f wmma16(v16h a, v16h b, v8f c) {
  return __builtin_amdgcn_wmma_f32_16x16x32_f16(false, a, false, b, (short)0, c, false, false);
}

// A fragment 16x32 f16 from LDS tile [row][k]; contiguous-K A layout:
// lane half h: vgpr 0..3 -> k = 16h+2v, vgpr 4..7 -> k = 16h+8+2(v-4)
__device__ __forceinline__ v16h load_frag_a(const f16* base, int ld, int group, int lane) {
  int m = lane & 15, h = lane >> 4;
  const f16* p = base + (group * 16 + m) * ld;
  v16h f;
#pragma unroll
  for (int v = 0; v < 4; ++v) {
    int k0 = h * 16 + 2 * v;
    f[2 * v]     = p[k0];
    f[2 * v + 1] = p[k0 + 1];
    int k1 = 8 + h * 16 + 2 * v;
    f[8 + 2 * v]     = p[k1];
    f[8 + 2 * v + 1] = p[k1 + 1];
  }
  return f;
}
// B fragment 32x16 f16 from LDS tile [n][k]; contiguous-K per half (k = 16h + v, same order as load_frag_a)
__device__ __forceinline__ v16h load_frag_b(const f16* base, int ld, int group, int lane) {
  int n = lane & 15, h = lane >> 4;
  const f16* p = base + (group * 16 + n) * ld + 16 * h;
  v16h f;
#pragma unroll
  for (int v = 0; v < 16; ++v) f[v] = p[v];
  return f;
}
// stage weight chunk W[kOff..kOff+32)[0..128) (row-major [K][128]) -> lds [n][k]
__device__ __forceinline__ void stage_b(f16* dst, const float* W, int kOff, int tid) {
  for (int idx = tid; idx < 32 * 128; idx += 256) {
    int k = idx >> 7, n = idx & 127;
    dst[n * LDB + k] = (f16)W[(size_t)(kOff + k) * 128 + n];
  }
}
// stage activation chunk src[rowBase..+128)[kOff..+32) f32 -> lds [r][k] f16 (guarded)
__device__ __forceinline__ void stage_a(f16* dst, const float* src, int rowBase, int nRows,
                                        int kOff, int tid) {
  for (int idx = tid; idx < 128 * 32; idx += 256) {
    int r = idx >> 5, k = idx & 31;
    int row = rowBase + r;
    float v = (row < nRows) ? src[(size_t)row * 128 + kOff + k] : 0.f;
    dst[r * LDA + k] = (f16)v;
  }
}

// ---------------- x init: x = ele_emb[z] + res_emb[z_res] ----------------
__global__ void __launch_bounds__(256) k_init_x(const int* __restrict__ z, const int* __restrict__ zr,
                                                const float* __restrict__ ele, const float* __restrict__ res,
                                                float* __restrict__ x, int N) {
  int t = blockIdx.x * 256 + threadIdx.x;
  if (t >= N * 128) return;
  int n = t >> 7, b = t & 127;
  x[t] = ele[z[n] * 128 + b] + res[zr[n] * 128 + b];
}

// -------- edge precompute: distances, cosine cutoff, RBF(+edge_emb) as f16 --------
__global__ void __launch_bounds__(256) k_edge_pre(const float* __restrict__ pos,
                                                  const int* __restrict__ idx_i, const int* __restrict__ idx_j,
                                                  const int* __restrict__ eattr, const float* __restrict__ eemb,
                                                  f16* __restrict__ fij, float* __restrict__ rcut, int E) {
  int e = blockIdx.x * 256 + threadIdx.x;
  if (e >= E) return;
  int i = idx_i[e], j = idx_j[e];
  float dx = pos[i * 3 + 0] - pos[j * 3 + 0];
  float dy = pos[i * 3 + 1] - pos[j * 3 + 1];
  float dz = pos[i * 3 + 2] - pos[j * 3 + 2];
  float d = sqrtf(dx * dx + dy * dy + dz * dz);
  const float CUT = 10.f;
  float rc = 0.5f * (__cosf(d * 3.14159265358979f / CUT) + 1.f);
  rcut[e] = (d < CUT) ? rc : 0.f;
  const float* ee = eemb + eattr[e] * 32;
  const float width = CUT / 31.f;              // linspace(0,10,32) spacing
  const float coeff = -0.5f / (width * width);
  f16* fo = fij + (size_t)e * 32;
#pragma unroll
  for (int k = 0; k < 32; ++k) {
    float t = d - width * (float)k;
    fo[k] = (f16)(__expf(coeff * t * t) + ee[k]);
  }
}

// ---------------- xf = x @ W  ([N,128]@[128,128]) ----------------
__global__ void __launch_bounds__(256) k_in2f(const float* __restrict__ x, const float* __restrict__ W,
                                              float* __restrict__ xf, int nRows) {
  extern __shared__ char smem[];
  f16* aT = (f16*)smem;          // 128*LDA
  f16* bT = aT + 128 * LDA;      // 128*LDB
  int tid = threadIdx.x, lane = tid & 31, wave = tid >> 5;
  int rowBase = blockIdx.x * 128;
  v8f acc[8];
#pragma unroll
  for (int t = 0; t < 8; ++t)
#pragma unroll
    for (int i = 0; i < 8; ++i) acc[t][i] = 0.f;

  for (int kc = 0; kc < 4; ++kc) {
    __syncthreads();
    stage_a(aT, x, rowBase, nRows, kc * 32, tid);
    stage_b(bT, W, kc * 32, tid);
    __syncthreads();
    v16h af = load_frag_a(aT, LDA, wave, lane);
#pragma unroll
    for (int t = 0; t < 8; ++t) acc[t] = wmma16(af, load_frag_b(bT, LDB, t, lane), acc[t]);
  }
  int h = lane >> 4, n15 = lane & 15;
#pragma unroll
  for (int t = 0; t < 8; ++t)
#pragma unroll
    for (int r = 0; r < 8; ++r) {
      int row = rowBase + wave * 16 + r + 8 * h;
      if (row < nRows) xf[(size_t)row * 128 + t * 16 + n15] = acc[t][r];
    }
}

// ---- fused edge filter: Wij = (ssp(f@W1+b1)@W2+b2)*rcut; scatter xf[j]*Wij into agg[i] ----
__global__ void __launch_bounds__(256) k_edge(const f16* __restrict__ fij, const float* __restrict__ rcut,
                                              const int* __restrict__ idx_i, const int* __restrict__ idx_j,
                                              const float* __restrict__ W1, const float* __restrict__ b1,
                                              const float* __restrict__ W2, const float* __restrict__ b2,
                                              const float* __restrict__ xf, float* __restrict__ agg, int nE) {
  extern __shared__ char smem[];
  f16* fT = (f16*)smem;                  // 128*LDA
  f16* wT = fT + 128 * LDA;              // 128*LDB (reused for W1 then W2 chunks)
  f16* hT = wT + 128 * LDB;              // 128*LDH
  float* wij = (float*)(hT + 128 * LDH); // 128*128 f32
  int tid = threadIdx.x, lane = tid & 31, wave = tid >> 5;
  int eBase = blockIdx.x * 128;
  int h = lane >> 4, n15 = lane & 15;

  for (int idx = tid; idx < 128 * 32; idx += 256) {
    int r = idx >> 5, k = idx & 31;
    int e = eBase + r;
    fT[r * LDA + k] = (e < nE) ? fij[(size_t)e * 32 + k] : (f16)0.f;
  }
  stage_b(wT, W1, 0, tid);
  __syncthreads();

  // GEMM1: [128,32] @ [32,128]
  v8f acc[8];
#pragma unroll
  for (int t = 0; t < 8; ++t)
#pragma unroll
    for (int i = 0; i < 8; ++i) acc[t][i] = 0.f;
  {
    v16h af = load_frag_a(fT, LDA, wave, lane);
#pragma unroll
    for (int t = 0; t < 8; ++t) acc[t] = wmma16(af, load_frag_b(wT, LDB, t, lane), acc[t]);
  }
  // ssp epilogue -> hT (f16, A-layout for GEMM2)
#pragma unroll
  for (int t = 0; t < 8; ++t) {
    int col = t * 16 + n15;
    float bb = b1[col];
#pragma unroll
    for (int r = 0; r < 8; ++r) {
      int row = wave * 16 + r + 8 * h;
      hT[row * LDH + col] = (f16)sspf(acc[t][r] + bb);
    }
  }

  // GEMM2: [128,128] @ [128,128]
  v8f acc2[8];
#pragma unroll
  for (int t = 0; t < 8; ++t)
#pragma unroll
    for (int i = 0; i < 8; ++i) acc2[t][i] = 0.f;
  for (int kc = 0; kc < 4; ++kc) {
    __syncthreads();                     // hT complete / wT free
    stage_b(wT, W2, kc * 32, tid);
    __syncthreads();
    v16h a2 = load_frag_a(hT + kc * 32, LDH, wave, lane);
#pragma unroll
    for (int t = 0; t < 8; ++t) acc2[t] = wmma16(a2, load_frag_b(wT, LDB, t, lane), acc2[t]);
  }
  // +b2, *rcut -> wij (LDS)
  float rcv[8];
#pragma unroll
  for (int r = 0; r < 8; ++r) {
    int e = eBase + wave * 16 + r + 8 * h;
    rcv[r] = (e < nE) ? rcut[e] : 0.f;
  }
#pragma unroll
  for (int t = 0; t < 8; ++t) {
    int col = t * 16 + n15;
    float bb = b2[col];
#pragma unroll
    for (int r = 0; r < 8; ++r) {
      int row = wave * 16 + r + 8 * h;
      wij[row * 128 + col] = (acc2[t][r] + bb) * rcv[r];
    }
  }
  __syncthreads();

  // gather xf[idx_j], multiply, scatter-add agg[idx_i] (coalesced, HW f32 atomics -> L2)
  for (int e = wave * 16; e < wave * 16 + 16; ++e) {
    int ge = eBase + e;
    if (ge >= nE) break;
    int j = idx_j[ge], i = idx_i[ge];
    const float* xfj = xf + (size_t)j * 128;
    float* aggi = agg + (size_t)i * 128;
#pragma unroll
    for (int c0 = 0; c0 < 128; c0 += 32) {
      int c = c0 + lane;
      unsafeAtomicAdd(&aggi[c], wij[e * 128 + c] * xfj[c]);
    }
  }
}

// ---------------- x += ssp(agg@W1+b1)@W2 + b2 ----------------
__global__ void __launch_bounds__(256) k_f2out(const float* __restrict__ agg,
                                               const float* __restrict__ W1, const float* __restrict__ b1,
                                               const float* __restrict__ W2, const float* __restrict__ b2,
                                               float* __restrict__ x, int nRows) {
  extern __shared__ char smem[];
  f16* aT = (f16*)smem;
  f16* wT = aT + 128 * LDA;
  f16* hT = wT + 128 * LDB;
  int tid = threadIdx.x, lane = tid & 31, wave = tid >> 5;
  int rowBase = blockIdx.x * 128;
  int h = lane >> 4, n15 = lane & 15;

  v8f acc[8];
#pragma unroll
  for (int t = 0; t < 8; ++t)
#pragma unroll
    for (int i = 0; i < 8; ++i) acc[t][i] = 0.f;
  for (int kc = 0; kc < 4; ++kc) {
    __syncthreads();
    stage_a(aT, agg, rowBase, nRows, kc * 32, tid);
    stage_b(wT, W1, kc * 32, tid);
    __syncthreads();
    v16h af = load_frag_a(aT, LDA, wave, lane);
#pragma unroll
    for (int t = 0; t < 8; ++t) acc[t] = wmma16(af, load_frag_b(wT, LDB, t, lane), acc[t]);
  }
#pragma unroll
  for (int t = 0; t < 8; ++t) {
    int col = t * 16 + n15;
    float bb = b1[col];
#pragma unroll
    for (int r = 0; r < 8; ++r) {
      int row = wave * 16 + r + 8 * h;
      hT[row * LDH + col] = (f16)sspf(acc[t][r] + bb);
    }
  }
  v8f acc2[8];
#pragma unroll
  for (int t = 0; t < 8; ++t)
#pragma unroll
    for (int i = 0; i < 8; ++i) acc2[t][i] = 0.f;
  for (int kc = 0; kc < 4; ++kc) {
    __syncthreads();
    stage_b(wT, W2, kc * 32, tid);
    __syncthreads();
    v16h a2 = load_frag_a(hT + kc * 32, LDH, wave, lane);
#pragma unroll
    for (int t = 0; t < 8; ++t) acc2[t] = wmma16(a2, load_frag_b(wT, LDB, t, lane), acc2[t]);
  }
#pragma unroll
  for (int t = 0; t < 8; ++t) {
    int col = t * 16 + n15;
    float bb = b2[col];
#pragma unroll
    for (int r = 0; r < 8; ++r) {
      int row = rowBase + wave * 16 + r + 8 * h;
      if (row < nRows) x[(size_t)row * 128 + col] += acc2[t][r] + bb;
    }
  }
}

// ---- head: h=silu(x/||x||); s=silu(h@W1+b1); y=s@W2; segment-sum over batch ----
__global__ void __launch_bounds__(256) k_head(const float* __restrict__ x, const int* __restrict__ batch,
                                              const float* __restrict__ W1, const float* __restrict__ b1,
                                              const float* __restrict__ W2,
                                              float* __restrict__ out, int nRows) {
  extern __shared__ char smem[];
  f16* aT = (f16*)smem;                   // 128*LDH (full-K silu(normalized x))
  f16* wT = aT + 128 * LDH;               // 128*LDB
  float* red = (float*)(wT + 128 * LDB);  // 128*16
  int tid = threadIdx.x, lane = tid & 31, wave = tid >> 5;
  int rowBase = blockIdx.x * 128;
  int h = lane >> 4, n15 = lane & 15;

  if (tid < 128) {
    int g = rowBase + tid;
    if (g < nRows) {
      const float* xr = x + (size_t)g * 128;
      float s2 = 0.f;
      for (int k = 0; k < 128; ++k) { float v = xr[k]; s2 += v * v; }
      float inv = 1.f / fmaxf(sqrtf(s2), 1e-12f);
      for (int k = 0; k < 128; ++k) aT[tid * LDH + k] = (f16)siluf(xr[k] * inv);
    } else {
      for (int k = 0; k < 128; ++k) aT[tid * LDH + k] = (f16)0.f;
    }
  }
  __syncthreads();

  v8f acc[8];
#pragma unroll
  for (int t = 0; t < 8; ++t)
#pragma unroll
    for (int i = 0; i < 8; ++i) acc[t][i] = 0.f;
  for (int kc = 0; kc < 4; ++kc) {
    __syncthreads();
    stage_b(wT, W1, kc * 32, tid);
    __syncthreads();
    v16h af = load_frag_a(aT + kc * 32, LDH, wave, lane);
#pragma unroll
    for (int t = 0; t < 8; ++t) acc[t] = wmma16(af, load_frag_b(wT, LDB, t, lane), acc[t]);
  }
  float part[8];
#pragma unroll
  for (int r = 0; r < 8; ++r) part[r] = 0.f;
#pragma unroll
  for (int t = 0; t < 8; ++t) {
    int col = t * 16 + n15;
    float bb = b1[col], w2 = W2[col];
#pragma unroll
    for (int r = 0; r < 8; ++r) part[r] += siluf(acc[t][r] + bb) * w2;
  }
#pragma unroll
  for (int r = 0; r < 8; ++r) {
    int row = wave * 16 + r + 8 * h;
    red[row * 16 + n15] = part[r];
  }
  __syncthreads();
  if (tid < 128) {
    float s = 0.f;
    for (int c = 0; c < 16; ++c) s += red[tid * 16 + c];
    int g = rowBase + tid;
    if (g < nRows) unsafeAtomicAdd(&out[batch[g]], s);
  }
}

extern "C" void kernel_launch(void* const* d_in, const int* in_sizes, int n_in,
                              void* d_out, int out_size, void* d_ws, size_t ws_size,
                              hipStream_t stream) {
  const int*   z      = (const int*)d_in[0];
  const int*   zres   = (const int*)d_in[1];
  const float* pos    = (const float*)d_in[2];
  const int*   idx_i  = (const int*)d_in[3];
  const int*   idx_j  = (const int*)d_in[4];
  const int*   eattr  = (const int*)d_in[5];
  const int*   batch  = (const int*)d_in[6];
  const float* ele    = (const float*)d_in[7];
  const float* res    = (const float*)d_in[8];
  const float* eemb   = (const float*)d_in[9];
  const float* in2fW  = (const float*)d_in[10];
  const float* fW1    = (const float*)d_in[11];
  const float* fb1    = (const float*)d_in[12];
  const float* fW2    = (const float*)d_in[13];
  const float* fb2    = (const float*)d_in[14];
  const float* oW1    = (const float*)d_in[15];
  const float* ob1    = (const float*)d_in[16];
  const float* oW2    = (const float*)d_in[17];
  const float* ob2    = (const float*)d_in[18];
  const float* hW1    = (const float*)d_in[19];
  const float* hb1    = (const float*)d_in[20];
  const float* hW2    = (const float*)d_in[21];
  (void)n_in; (void)ws_size;

  const int N = in_sizes[0];
  const int E = in_sizes[3];

  size_t off = 0;
  auto take = [&](size_t bytes) {
    char* p = (char*)d_ws + off;
    off += (bytes + 255) & ~(size_t)255;
    return (void*)p;
  };
  float* x    = (float*)take((size_t)N * 128 * 4);
  float* xf   = (float*)take((size_t)N * 128 * 4);
  float* agg  = (float*)take((size_t)N * 128 * 4);
  float* rcut = (float*)take((size_t)E * 4);
  f16*   fij  = (f16*)take((size_t)E * 32 * 2);

  hipMemsetAsync(d_out, 0, (size_t)out_size * sizeof(float), stream);
  k_init_x<<<dim3((unsigned)((N * 128 + 255) / 256)), dim3(256), 0, stream>>>(z, zres, ele, res, x, N);
  k_edge_pre<<<dim3((unsigned)((E + 255) / 256)), dim3(256), 0, stream>>>(pos, idx_i, idx_j, eattr, eemb, fij, rcut, E);

  const unsigned rowBlocks  = (unsigned)((N + 127) / 128);
  const unsigned edgeBlocks = (unsigned)((E + 127) / 128);
  const size_t shGemm = (size_t)(128 * LDA + 128 * LDB) * 2;
  const size_t shEdge = (size_t)(128 * LDA + 128 * LDB + 128 * LDH) * 2 + (size_t)128 * 128 * 4;
  const size_t shF2   = (size_t)(128 * LDA + 128 * LDB + 128 * LDH) * 2;
  const size_t shHead = (size_t)(128 * LDH + 128 * LDB) * 2 + (size_t)128 * 16 * 4;

  for (int it = 0; it < 6; ++it) {
    k_in2f<<<dim3(rowBlocks), dim3(256), shGemm, stream>>>(x, in2fW + (size_t)it * 128 * 128, xf, N);
    hipMemsetAsync(agg, 0, (size_t)N * 128 * 4, stream);
    k_edge<<<dim3(edgeBlocks), dim3(256), shEdge, stream>>>(
        fij, rcut, idx_i, idx_j,
        fW1 + (size_t)it * 32 * 128, fb1 + (size_t)it * 128,
        fW2 + (size_t)it * 128 * 128, fb2 + (size_t)it * 128, xf, agg, E);
    k_f2out<<<dim3(rowBlocks), dim3(256), shF2, stream>>>(
        agg, oW1 + (size_t)it * 128 * 128, ob1 + (size_t)it * 128,
        oW2 + (size_t)it * 128 * 128, ob2 + (size_t)it * 128, x, N);
  }
  k_head<<<dim3(rowBlocks), dim3(256), shHead, stream>>>(x, batch, hW1, hb1, hW2, (float*)d_out, N);
}
